// SelfModifyingLayer_7275674599447
// MI455X (gfx1250) — compile-verified, matrix operand order checked by reading, not run on hardware
//
#include <hip/hip_runtime.h>
#include <hip/hip_bf16.h>
#include <math.h>

// Problem dims (compile-time; match reference)
static constexpr int DIM   = 1024;
static constexpr int BATCH = 8;
static constexpr int SEQ   = 2048;
static constexpr int MTOT  = BATCH * SEQ;   // 16384 rows
static constexpr int CHUNK = 64;            // time chunk
static constexpr int ETILE = 128;           // value-dim tile per workgroup

typedef __attribute__((ext_vector_type(16))) __bf16 v16bf;
typedef __attribute__((ext_vector_type(8)))  float  v8f;
typedef int b128_t __attribute__((vector_size(16)));   // 4 x i32, matches builtin param

union Frag16 {
  v16bf v;
  unsigned short u[16];
};
union U4 {
  uint4 v;
  unsigned short h[8];
};

// -------- CDNA5 async global->LDS copy (probe-verified on this toolchain) --
#if defined(__AMDGCN__) && __has_builtin(__builtin_amdgcn_global_load_async_to_lds_b128) && __has_builtin(__builtin_amdgcn_s_wait_asynccnt)
#define HAVE_ASYNC_LDS 1
#endif

__device__ __forceinline__ void cp16(void* dst_lds, const void* src_g) {
#ifdef HAVE_ASYNC_LDS
  __builtin_amdgcn_global_load_async_to_lds_b128(
      (__attribute__((address_space(1))) b128_t*)(void*)src_g,
      (__attribute__((address_space(3))) b128_t*)dst_lds, 0, 0);
#else
  *(uint4*)dst_lds = *(const uint4*)src_g;
#endif
}
__device__ __forceinline__ void cp_wait() {
#ifdef HAVE_ASYNC_LDS
  __builtin_amdgcn_s_wait_asynccnt(0);
#endif
}

// ---------------------------- math helpers ---------------------------------
__device__ __forceinline__ float fast_exp2(float x) {
#if defined(__AMDGCN__) && __has_builtin(__builtin_amdgcn_exp2f)
  return __builtin_amdgcn_exp2f(x);   // v_exp_f32
#else
  return exp2f(x);
#endif
}

// ---------------------------- bf16 helpers ---------------------------------
__device__ __forceinline__ unsigned short f2bf(float f) {
  unsigned int x = __float_as_uint(f);
  unsigned int r = x + 0x7fffu + ((x >> 16) & 1u);
  return (unsigned short)(r >> 16);
}
__device__ __forceinline__ float bf2f(unsigned short u) {
  return __uint_as_float(((unsigned int)u) << 16);
}
#if defined(__AMDGCN__) && __has_builtin(__builtin_amdgcn_cvt_pk_bf16_f32)
typedef __attribute__((ext_vector_type(2))) __bf16 v2bf;
__device__ __forceinline__ unsigned pk2(float a, float b) {  // v_cvt_pk_bf16_f32
  union { v2bf v; unsigned u; } c;
  c.v = __builtin_amdgcn_cvt_pk_bf16_f32(a, b);
  return c.u;
}
#else
__device__ __forceinline__ unsigned pk2(float a, float b) {
  return (unsigned)f2bf(a) | ((unsigned)f2bf(b) << 16);
}
#endif

// ------------------------ WMMA fragment helpers ----------------------------
// CDNA5 16-bit A-matrix 16x32 per-lane layout: element e (0..15) of the
// v16bf fragment maps to K = (e<8 ? hi*8+e : 16+hi*8+(e-8)), hi = lane>>4.
__device__ __forceinline__ int frag_k(int e, int hi) {
  return ((e >> 3) << 4) + hi * 8 + (e & 7);
}
__device__ __forceinline__ Frag16 load_a_frag(const unsigned short* lds,
                                              int row0, int col0, int stride) {
  int lane = threadIdx.x & 31;
  int m = lane & 15, hi = lane >> 4;
  Frag16 f;
#pragma unroll
  for (int e = 0; e < 16; ++e)
    f.u[e] = lds[(row0 + m) * stride + col0 + frag_k(e, hi)];
  return f;
}
__device__ __forceinline__ Frag16 load_b_frag(const unsigned short* lds,
                                              int k0, int n0, int stride) {
  int lane = threadIdx.x & 31;
  int n = lane & 15, hi = lane >> 4;
  Frag16 f;
#pragma unroll
  for (int e = 0; e < 16; ++e)
    f.u[e] = lds[(k0 + frag_k(e, hi)) * stride + n0 + n];
  return f;
}
__device__ __forceinline__ Frag16 load_bT_frag(const unsigned short* lds,
                                               int n0, int k0, int stride) {
  int lane = threadIdx.x & 31;
  int n = lane & 15, hi = lane >> 4;
  Frag16 f;
#pragma unroll
  for (int e = 0; e < 16; ++e)
    f.u[e] = lds[(n0 + n) * stride + k0 + frag_k(e, hi)];
  return f;
}
__device__ __forceinline__ v8f wmma_bf16(const Frag16& a, const Frag16& b, v8f c) {
  return __builtin_amdgcn_wmma_f32_16x16x32_bf16(false, a.v, false, b.v,
                                                 (short)0, c, false, false);
}

// ---------------------------------------------------------------------------
// Tiled GEMM: Y[M,N] = act(X[M,K] @ W[K,N] + bias). ACT: 0=none, 1=elu+1.
// 128x128 block tile, 64-deep k-steps, 256 threads = 8 waves.
// ---------------------------------------------------------------------------
template <int ACT, bool IN_BF16, bool OUT_BF16>
__global__ void __launch_bounds__(256)
gemm_bias_act(const void* __restrict__ Xv, const float* __restrict__ W,
              const float* __restrict__ bias, void* __restrict__ Yv,
              int Mdim, int Ndim, int Kdim) {
  __shared__ __attribute__((aligned(16))) unsigned short sX[128][72];
  __shared__ __attribute__((aligned(16))) unsigned short sW[64][136];

  const int tid = threadIdx.x;
  const int wv  = tid >> 5;
  const int m0  = blockIdx.y * 128;
  const int n0  = blockIdx.x * 128;

  v8f acc[8];
#pragma unroll
  for (int t = 0; t < 8; ++t) acc[t] = {};

  const float*          Xf = (const float*)Xv;
  const unsigned short* Xb = (const unsigned short*)Xv;

  for (int k0 = 0; k0 < Kdim; k0 += 64) {
    __syncthreads();
    if (IN_BF16) {
      // 128x64 bf16 = 1024 x 16B, async DMA to LDS
#pragma unroll
      for (int i = tid; i < 1024; i += 256) {
        int r = i >> 3, c = (i & 7) * 8;
        cp16(&sX[r][c], Xb + (size_t)(m0 + r) * Kdim + k0 + c);
      }
    } else {
      // 128x64 f32: float4 load -> 4xbf16 packed (uint2) store
#pragma unroll
      for (int i = tid; i < 2048; i += 256) {
        int r = i >> 4, c = (i & 15) * 4;
        float4 f = *(const float4*)(Xf + (size_t)(m0 + r) * Kdim + k0 + c);
        uint2 p; p.x = pk2(f.x, f.y); p.y = pk2(f.z, f.w);
        *(uint2*)&sX[r][c] = p;
      }
    }
    // W tile 64x128 f32 -> bf16
#pragma unroll
    for (int i = tid; i < 2048; i += 256) {
      int r = i >> 5, c = (i & 31) * 4;
      float4 f = *(const float4*)(W + (size_t)(k0 + r) * Ndim + n0 + c);
      uint2 p; p.x = pk2(f.x, f.y); p.y = pk2(f.z, f.w);
      *(uint2*)&sW[r][c] = p;
    }
    if (k0 + 64 < Kdim) {
      const char* base = IN_BF16
          ? (const char*)&Xb[(size_t)(m0 + (tid >> 1)) * Kdim + k0 + 64]
          : (const char*)&Xf[(size_t)(m0 + (tid >> 1)) * Kdim + k0 + 64];
      __builtin_prefetch(base, 0, 1);   // global_prefetch_b8
    }
    cp_wait();
    __syncthreads();

    Frag16 a0 = load_a_frag(&sX[0][0], wv * 16, 0, 72);
    Frag16 a1 = load_a_frag(&sX[0][0], wv * 16, 32, 72);
#pragma unroll
    for (int t = 0; t < 8; ++t) {
      Frag16 b0 = load_b_frag(&sW[0][0], 0, t * 16, 136);
      acc[t] = wmma_bf16(a0, b0, acc[t]);
      Frag16 b1 = load_b_frag(&sW[0][0], 32, t * 16, 136);
      acc[t] = wmma_bf16(a1, b1, acc[t]);
    }
  }

  const int lane = tid & 31;
  const int hi = lane >> 4, nn = lane & 15;
#pragma unroll
  for (int t = 0; t < 8; ++t) {
    int col = n0 + t * 16 + nn;
    float bval = bias[col];
#pragma unroll
    for (int r = 0; r < 8; ++r) {
      int row = m0 + wv * 16 + hi * 8 + r;
      float y = acc[t][r] + bval;
      if (ACT == 1) y = (y > 0.f) ? (y + 1.f) : __expf(y);
      size_t g = (size_t)row * Ndim + col;
      if (OUT_BF16) ((unsigned short*)Yv)[g] = f2bf(y);
      else          ((float*)Yv)[g] = y;
    }
  }
}

__global__ void __launch_bounds__(256)
memzero_f32v4(float4* __restrict__ p, size_t n4) {
  size_t i = (size_t)blockIdx.x * blockDim.x + threadIdx.x;
  size_t stride = (size_t)gridDim.x * blockDim.x;
  float4 z; z.x = z.y = z.z = z.w = 0.f;
  for (; i < n4; i += stride) p[i] = z;
}

// ---------------------------------------------------------------------------
// Chunked linear attention. Grid = (DIM/ETILE, BATCH). Each workgroup owns a
// [DIM x ETILE] slice of the recurrent state (32MB total -> L2-resident on
// MI455X's 192MB L2) and walks the 32 chunks sequentially.
// ---------------------------------------------------------------------------
__global__ void __launch_bounds__(256)
linattn_chunk(const unsigned short* __restrict__ qb,
              const unsigned short* __restrict__ kb,
              const unsigned short* __restrict__ vb,
              float* __restrict__ mem,
              unsigned short* __restrict__ attn,
              const float* __restrict__ decay_p) {
  __shared__ __attribute__((aligned(16))) unsigned short s_q[64][136];  // A: Q slab / B: scaled Q
  __shared__ __attribute__((aligned(16))) unsigned short s_k[64][136];  // A: K slab
  __shared__ __attribute__((aligned(16))) unsigned short s_p[64][72];   // masked scores
  __shared__ __attribute__((aligned(16))) unsigned short s_v[64][136];  // V tile
  __shared__ __attribute__((aligned(16))) unsigned short s_m[64][136];  // Mem slab (bf16)
  __shared__ __attribute__((aligned(16))) unsigned short s_kt[128][72]; // scaled K^T block
  __shared__ float s_dpow[64];

  const int tid  = threadIdx.x;
  const int wv   = tid >> 5;
  const int lane = tid & 31;
  const int hi   = lane >> 4, nn = lane & 15;

  const int b  = blockIdx.y;
  const int e0 = blockIdx.x * ETILE;

  const float pr    = decay_p[0];
  const float decay = 1.f / (1.f + __expf(-pr));
  const float l2d   = __log2f(decay);
  const float dec64 = __powf(decay, 64.f);

  if (tid < 64) s_dpow[tid] = __powf(decay, (float)tid);
  __syncthreads();

  const unsigned short* q_b = qb + (size_t)b * SEQ * DIM;
  const unsigned short* k_b = kb + (size_t)b * SEQ * DIM;
  const unsigned short* v_b = vb + (size_t)b * SEQ * DIM;
  float*          mem_b = mem  + (size_t)b * DIM * DIM;   // [d][e]
  unsigned short* o_b   = attn + (size_t)b * SEQ * DIM;

  for (int c = 0; c < SEQ / CHUNK; ++c) {
    const int t0 = c * CHUNK;

    // ---------- Phase A: S = Q Kc^T, masked with decay^(t-1-s) ----------
    const int rt  = wv >> 1;
    const int ctb = (wv & 1) * 2;
    v8f accs[2]; accs[0] = {}; accs[1] = {};
    for (int k0 = 0; k0 < DIM; k0 += 128) {
      __syncthreads();
      // 64x128 bf16 per matrix = 1024 x 16B async each
#pragma unroll
      for (int i = tid; i < 1024; i += 256) {
        int r = i >> 4, cc = (i & 15) * 8;
        size_t g = (size_t)(t0 + r) * DIM + k0 + cc;
        cp16(&s_q[r][cc], q_b + g);
        cp16(&s_k[r][cc], k_b + g);
      }
      cp_wait();
      __syncthreads();
#pragma unroll
      for (int kk = 0; kk < 128; kk += 32) {
        Frag16 a = load_a_frag(&s_q[0][0], rt * 16, kk, 136);
#pragma unroll
        for (int j = 0; j < 2; ++j) {
          Frag16 bb = load_bT_frag(&s_k[0][0], (ctb + j) * 16, kk, 136);
          accs[j] = wmma_bf16(a, bb, accs[j]);
        }
      }
    }
    __syncthreads();
    // branch-free mask + decay -> P (weight via exp2, select via cndmask)
#pragma unroll
    for (int j = 0; j < 2; ++j) {
      int nc = (ctb + j) * 16 + nn;
#pragma unroll
      for (int r = 0; r < 8; ++r) {
        int mr = rt * 16 + hi * 8 + r;
        int d  = mr - 1 - nc;
        float w  = fast_exp2((float)d * l2d);
        float pv = (d >= 0) ? accs[j][r] * w : 0.f;
        s_p[mr][nc] = f2bf(pv);
      }
    }
    // stage V tile 64x128 bf16 = 1024 x 16B async
#pragma unroll
    for (int i = tid; i < 1024; i += 256) {
      int r = i >> 4, cc = (i & 15) * 8;
      cp16(&s_v[r][cc], v_b + (size_t)(t0 + r) * DIM + e0 + cc);
    }
    cp_wait();
    __syncthreads();

    // ---------- Phase B: O = P@V + diag(decay^tau) Q @ Mem ----------
    const int orow = (wv >> 1) * 16;
    const int ocol = (wv & 1) * 64;
    v8f acco[4];
#pragma unroll
    for (int j = 0; j < 4; ++j) acco[j] = {};
    // intra contribution (K = 64)
#pragma unroll
    for (int k0 = 0; k0 < 64; k0 += 32) {
      Frag16 a = load_a_frag(&s_p[0][0], orow, k0, 72);
#pragma unroll
      for (int j = 0; j < 4; ++j) {
        Frag16 bb = load_b_frag(&s_v[0][0], k0, ocol + j * 16, 136);
        acco[j] = wmma_bf16(a, bb, acco[j]);
      }
    }
    // inter contribution, 64-deep k-steps over L2-resident Mem
    for (int k0 = 0; k0 < DIM; k0 += 64) {
      __syncthreads();
      // decay-scaled Q slab 64x64: uint4 load, unpack/scale/repack
#pragma unroll
      for (int i = tid; i < 512; i += 256) {
        int r = i >> 3, cc = (i & 7) * 8;
        U4 u; u.v = *(const uint4*)(q_b + (size_t)(t0 + r) * DIM + k0 + cc);
        float s = s_dpow[r];
        U4 o;
#pragma unroll
        for (int j = 0; j < 4; ++j)
          ((unsigned*)&o.v)[j] = pk2(bf2f(u.h[2 * j]) * s, bf2f(u.h[2 * j + 1]) * s);
        *(uint4*)&s_q[r][cc] = o.v;
      }
      // Mem slab 64x128 f32 -> bf16
#pragma unroll
      for (int i = tid; i < 2048; i += 256) {
        int r = i >> 5, cc = (i & 31) * 4;
        float4 f = *(const float4*)(mem_b + (size_t)(k0 + r) * DIM + e0 + cc);
        uint2 p; p.x = pk2(f.x, f.y); p.y = pk2(f.z, f.w);
        *(uint2*)&s_m[r][cc] = p;
      }
      __syncthreads();
#pragma unroll
      for (int kk = 0; kk < 64; kk += 32) {
        Frag16 a = load_a_frag(&s_q[0][0], orow, kk, 136);
#pragma unroll
        for (int j = 0; j < 4; ++j) {
          Frag16 bb = load_b_frag(&s_m[0][0], kk, ocol + j * 16, 136);
          acco[j] = wmma_bf16(a, bb, acco[j]);
        }
      }
    }
#pragma unroll
    for (int j = 0; j < 4; ++j) {
      int col = e0 + ocol + j * 16 + nn;
#pragma unroll
      for (int r = 0; r < 8; ++r) {
        int row = t0 + orow + hi * 8 + r;
        o_b[(size_t)row * DIM + col] = f2bf(acco[j][r]);
      }
    }

    // ---------- Phase C: Mem = dec64*Mem + Kscaled^T @ V ----------
    for (int db = 0; db < DIM / 128; ++db) {
      const int d0 = db * 128;
      __syncthreads();
      // stage transposed, decay-scaled K block: two tau rows x 8 features
      // per op; packed pair store along contiguous tau axis (b32)
#pragma unroll
      for (int i = tid; i < 512; i += 256) {
        int tau2 = (i >> 4) * 2, dcb = (i & 15) * 8;
        U4 ua, ub;
        ua.v = *(const uint4*)(k_b + (size_t)(t0 + tau2) * DIM + d0 + dcb);
        ub.v = *(const uint4*)(k_b + (size_t)(t0 + tau2 + 1) * DIM + d0 + dcb);
        float sa = s_dpow[63 - tau2], sb = s_dpow[62 - tau2];
#pragma unroll
        for (int j = 0; j < 8; ++j) {
          unsigned w = pk2(bf2f(ua.h[j]) * sa, bf2f(ub.h[j]) * sb);
          *(unsigned*)&s_kt[dcb + j][tau2] = w;
        }
      }
      __syncthreads();
      // hoisted A fragments (invariant over the 8 column tiles)
      Frag16 a0 = load_a_frag(&s_kt[0][0], wv * 16, 0, 72);
      Frag16 a1 = load_a_frag(&s_kt[0][0], wv * 16, 32, 72);
#pragma unroll
      for (int ct = 0; ct < 8; ++ct) {
        v8f accm = {};
        Frag16 b0 = load_b_frag(&s_v[0][0], 0, ct * 16, 136);
        Frag16 b1 = load_b_frag(&s_v[0][0], 32, ct * 16, 136);
        accm = wmma_bf16(a0, b0, accm);
        accm = wmma_bf16(a1, b1, accm);
        int col = e0 + ct * 16 + nn;
#pragma unroll
        for (int r = 0; r < 8; ++r) {
          int row = d0 + wv * 16 + hi * 8 + r;
          size_t idx = (size_t)row * DIM + col;
          mem_b[idx] = dec64 * mem_b[idx] + accm[r];
        }
      }
    }
    __threadfence();
    __syncthreads();
  }
}

// ---------------------------------------------------------------------------
// Workspace layout (bytes): q 32MB | k 32MB | v 32MB | mem 32MB | attn 32MB
// ---------------------------------------------------------------------------
extern "C" void kernel_launch(void* const* d_in, const int* in_sizes, int n_in,
                              void* d_out, int out_size, void* d_ws, size_t ws_size,
                              hipStream_t stream) {
  (void)in_sizes; (void)n_in; (void)out_size; (void)ws_size;

  const float* x     = (const float*)d_in[0];
  const float* Wq    = (const float*)d_in[1];
  const float* bq    = (const float*)d_in[2];
  const float* Wk    = (const float*)d_in[3];
  const float* bk    = (const float*)d_in[4];
  const float* Wv    = (const float*)d_in[5];
  const float* bv    = (const float*)d_in[6];
  const float* Wo    = (const float*)d_in[7];
  const float* bo    = (const float*)d_in[8];
  const float* decay = (const float*)d_in[9];
  float* out = (float*)d_out;

  char* ws = (char*)d_ws;
  const size_t bf16Plane = (size_t)MTOT * DIM * sizeof(unsigned short);  // 32MB
  unsigned short* q_ws   = (unsigned short*)(ws);
  unsigned short* k_ws   = (unsigned short*)(ws + bf16Plane);
  unsigned short* v_ws   = (unsigned short*)(ws + 2 * bf16Plane);
  float*          mem_ws = (float*)(ws + 3 * bf16Plane);
  unsigned short* o_ws   = (unsigned short*)(ws + 3 * bf16Plane +
                                             (size_t)BATCH * DIM * DIM * sizeof(float));

  dim3 gGemm(DIM / 128, MTOT / 128);  // (8, 128)
  dim3 blk(256);

  gemm_bias_act<0, false, true><<<gGemm, blk, 0, stream>>>(x, Wq, bq, q_ws, MTOT, DIM, DIM);
  gemm_bias_act<1, false, true><<<gGemm, blk, 0, stream>>>(x, Wk, bk, k_ws, MTOT, DIM, DIM);
  gemm_bias_act<0, false, true><<<gGemm, blk, 0, stream>>>(x, Wv, bv, v_ws, MTOT, DIM, DIM);

  memzero_f32v4<<<1024, blk, 0, stream>>>((float4*)mem_ws,
                                          (size_t)BATCH * DIM * DIM / 4);

  dim3 gAttn(DIM / ETILE, BATCH);  // (8, 8)
  linattn_chunk<<<gAttn, blk, 0, stream>>>(q_ws, k_ws, v_ws, mem_ws, o_ws, decay);

  gemm_bias_act<0, true, false><<<gGemm, blk, 0, stream>>>(o_ws, Wo, bo, out, MTOT, DIM, DIM);
}
